// GGNN_35871566856623
// MI455X (gfx1250) — matrix operand order repaired and checked by reading, not compiled, not run on hardware
//
#include <hip/hip_runtime.h>

// ---------------------------------------------------------------------------
// GGNN on MI455X (gfx1250): bf16 WMMA pipeline.
//   kA:  a_t[e][b][f][n] = (feats @ fc_w[e]^T + fc_b[e])   (transposed store)
//   kB:  flows[b][n][e][f] = g_bf[b][e] (NxN) @ a[e][b] (NxF)
//        - B panel (32KB contiguous) staged to LDS via TENSOR_LOAD_TO_LDS (TDM),
//          padded 16B/1KB by the TDM to avoid LDS bank conflicts.
//   kC:  fused GRU cell (two WMMA GEMM chains + gate math)
// ---------------------------------------------------------------------------

typedef __bf16 bf16_t;
typedef __attribute__((ext_vector_type(16))) __bf16 v16bf;
typedef __attribute__((ext_vector_type(8)))  __bf16 v8bf;
typedef __attribute__((ext_vector_type(8)))  float  v8f;
typedef __attribute__((ext_vector_type(4)))  unsigned int u32x4;
typedef __attribute__((ext_vector_type(8)))  int i32x8;
typedef __attribute__((ext_vector_type(4)))  int i32x4;

#define WMMA_BF16(a, b, c) \
  __builtin_amdgcn_wmma_f32_16x16x32_bf16(false, (a), false, (b), (short)0, (c), false, false)

// --- A fragment: 16x32 tile from row-major [ld] source ----------------------
// A 16x32 in this code's k order (same as B): lane<16 holds row=lane, K=0..15;
// lane>=16 holds row=lane-16, K=16..31. Two 16B loads per lane.
__device__ __forceinline__ v16bf load_a_frag(const bf16_t* base, int ld, int k0) {
  const int lane = threadIdx.x & 31;
  const int row  = lane & 15;
  const int koff = (lane >> 4) << 4;          // 0 or 16
  const bf16_t* p = base + (size_t)row * ld + k0 + koff;
  v8bf lo = *(const v8bf*)(p);
  v8bf hi = *(const v8bf*)(p + 8);
  v16bf a;
#pragma unroll
  for (int i = 0; i < 8; ++i) { a[i] = lo[i]; a[i + 8] = hi[i]; }
  return a;
}

// --- B fragment: 32x16 tile, source stored TRANSPOSED (N-major, [ldt]) ------
// lane<16 holds col=lane, K=0..15; lane>=16 holds col=lane-16, K=16..31.
__device__ __forceinline__ v16bf load_bt_frag(const bf16_t* baseT, int ldt, int k0) {
  const int lane = threadIdx.x & 31;
  const int col  = lane & 15;
  const int kb   = (lane >> 4) << 4;          // 0 or 16
  return *(const v16bf*)(baseT + (size_t)col * ldt + k0 + kb);
}

// --- B fragment from the TDM-staged LDS panel -------------------------------
// Panel layout after TDM pad: 1024B chunks at stride 1040B; column c (= panel
// row c, 2048B) occupies chunks 2c,2c+1 -> byte(c,m) = c*2080 + 2m + 16*(m>>9).
// k chunks are 32B-aligned and never straddle the 16B pad gap.
__device__ __forceinline__ v16bf load_bt_lds(const bf16_t* lds, int k0) {
  const int lane = threadIdx.x & 31;
  const int col  = lane & 15;
  const int m    = k0 + ((lane >> 4) << 4);
  const bf16_t* p = (const bf16_t*)((const char*)lds + col * 2080 + m * 2 + ((m >> 9) << 4));
  v8bf lo = *(const v8bf*)(p);
  v8bf hi = *(const v8bf*)(p + 8);
  v16bf b;
#pragma unroll
  for (int i = 0; i < 8; ++i) { b[i] = lo[i]; b[i + 8] = hi[i]; }
  return b;
}

__device__ __forceinline__ float sigmoidf(float x) { return 1.0f / (1.0f + __expf(-x)); }

// ---------------------------------------------------------------------------
// Prep kernels
// ---------------------------------------------------------------------------
__global__ void scale_graph_kernel(const int* __restrict__ graphs,
                                   const float* __restrict__ tw,
                                   bf16_t* __restrict__ g_bf, int n) {
  float sw[4];
#pragma unroll
  for (int e = 0; e < 4; ++e) sw[e] = 1.0f / (1.0f + __expf(-tw[e]));
  for (int i = blockIdx.x * blockDim.x + threadIdx.x; i < n; i += gridDim.x * blockDim.x) {
    const int e = (i >> 20) & 3;              // N*N = 2^20 elements per (b,e) slice
    g_bf[i] = (bf16_t)((float)graphs[i] * sw[e]);
  }
}

__global__ void cvt_f32_bf16_kernel(const float* __restrict__ src,
                                    bf16_t* __restrict__ dst, int n) {
  int i = blockIdx.x * blockDim.x + threadIdx.x;
  if (i < n) dst[i] = (bf16_t)src[i];
}

__global__ void cvt_copy_kernel(const float* __restrict__ src,
                                bf16_t* __restrict__ dbf,
                                float* __restrict__ dfp, int n) {
  int i = blockIdx.x * blockDim.x + threadIdx.x;
  if (i < n) { float v = src[i]; dbf[i] = (bf16_t)v; dfp[i] = v; }
}

// ---------------------------------------------------------------------------
// kA: a_t[e][b][g][n] = sum_f feats_bf[b][n][f] * fcw_bf[e][g][f] + fc_b[e][g]
// 16384 wave-tiles = 2048 blocks x 8 waves.
// ---------------------------------------------------------------------------
__global__ void kA(const bf16_t* __restrict__ feats_bf,
                   const bf16_t* __restrict__ fcw_bf,
                   const float*  __restrict__ fc_b,
                   bf16_t* __restrict__ a_t) {
  const int tile  = blockIdx.x * 8 + (threadIdx.x >> 5);
  const int ntile = tile & 63;
  const int gtile = (tile >> 6) & 7;
  const int eb    = tile >> 9;                // e*8 + b
  const int e     = eb >> 3;
  const int lane  = threadIdx.x & 31;
  const int col   = lane & 15;
  const int hi    = lane >> 4;

  const bf16_t* A  = feats_bf + (size_t)(eb & 7) * (1024 * 128) + (size_t)ntile * 16 * 128;
  const bf16_t* BT = fcw_bf + (size_t)e * (128 * 128) + (size_t)gtile * 16 * 128;

  const float bias = fc_b[e * 128 + gtile * 16 + col];
  v8f c;
#pragma unroll
  for (int v = 0; v < 8; ++v) c[v] = bias;

#pragma unroll
  for (int k = 0; k < 128; k += 32) {
    v16bf a  = load_a_frag(A, 128, k);
    v16bf bb = load_bt_frag(BT, 128, k);
    c = WMMA_BF16(a, bb, c);
  }

  bf16_t* out = a_t + (size_t)eb * (128 * 1024);
  const int g = gtile * 16 + col;
  const int n = ntile * 16 + hi * 8;
  v8bf d;
#pragma unroll
  for (int v = 0; v < 8; ++v) d[v] = (bf16_t)c[v];
  *(v8bf*)(out + (size_t)g * 1024 + n) = d;
}

// ---------------------------------------------------------------------------
// kB: flows[b][n][e][f] = sum_m g_bf[b][e][n][m] * a[e][b][m][f]
// Block = 8 waves sharing one 32KB a_t panel (16 f-rows x 1024 m), staged to
// LDS by the Tensor Data Mover with 16B pad per 1KB (bank spreading).
// Grid: be(32) x ftile(8) x nblk(8) = 2048 blocks of 256 threads.
// ---------------------------------------------------------------------------
__global__ void kB(const bf16_t* __restrict__ g_bf,
                   const bf16_t* __restrict__ a_t,
                   bf16_t* __restrict__ flows) {
  __shared__ bf16_t panel[33280 / 2];         // 32 x (1024B + 16B pad)

  const int nblk  = blockIdx.x & 7;
  const int ftile = (blockIdx.x >> 3) & 7;
  const int be    = blockIdx.x >> 6;          // b*4 + e
  const int b     = be >> 2;
  const int e     = be & 3;
  const int wid   = threadIdx.x >> 5;
  const int lane  = threadIdx.x & 31;
  const int col   = lane & 15;
  const int hi    = lane >> 4;

  // ---- TDM: stage the contiguous 32KB B panel into LDS (wave 0 only) ------
  if (wid == 0) {
    const unsigned long long ga =
        (unsigned long long)(const void*)(a_t + (size_t)(e * 8 + b) * (128 * 1024) +
                                          (size_t)ftile * 16 * 1024);
    const unsigned lds_off =
        (unsigned)(unsigned long long)(__attribute__((address_space(3))) bf16_t*)panel;
    // D# group0: count=1 | lds_addr | global_addr | type=2
    u32x4 g0 = {1u, lds_off, (unsigned)ga,
                (unsigned)((ga >> 32) & 0x01FFFFFFull) | 0x80000000u};
    // D# group1: data_size=3 (8B), pad_enable, pad_interval=7 (1024B),
    // pad_amount=3 (16B); tensor 128x32 units, tile 128x32, stride0=128.
    i32x8 g1 = {(int)((3u << 16) | (1u << 20) | (7u << 22) | (3u << 25)),
                (int)(128u << 16),            // tensor_dim0 = 128 (lo16 in [63:48])
                (int)(32u << 16),             // tensor_dim0 hi | tensor_dim1 = 32
                (int)(128u << 16),            // tensor_dim1 hi | tile_dim0 = 128
                32,                           // tile_dim1 = 32 | tile_dim2 = 0
                128,                          // tensor_dim0_stride lo
                0, 0};
    i32x4 z4 = {0, 0, 0, 0};
    i32x8 z8 = {0, 0, 0, 0, 0, 0, 0, 0};
    __builtin_amdgcn_tensor_load_to_lds(g0, g1, z4, z4, z8, 0);
    __builtin_amdgcn_s_wait_tensorcnt(0);
  }
  __syncthreads();

  // ---- compute: each wave one 16x16 tile, K = 1024 -> 32 WMMAs ------------
  const int ntile = nblk * 8 + wid;
  const bf16_t* A = g_bf + (size_t)be * (1024 * 1024) + (size_t)ntile * 16 * 1024;

  v8f c = {};
#pragma unroll 8
  for (int k = 0; k < 1024; k += 32) {
    v16bf a  = load_a_frag(A, 1024, k);
    v16bf bb = load_bt_lds(panel, k);
    c = WMMA_BF16(a, bb, c);
  }

  const int f = ftile * 16 + col;
  const size_t rowbase = (size_t)b * 1024 + ntile * 16 + hi * 8;
#pragma unroll
  for (int v = 0; v < 8; ++v)
    flows[(rowbase + v) * 512 + (size_t)e * 128 + f] = (bf16_t)c[v];
}

// ---------------------------------------------------------------------------
// kC: fused GRU cell on 16x16 tiles.  4096 wave-tiles = 512 blocks x 8 waves.
// ---------------------------------------------------------------------------
__global__ void kC(const bf16_t* __restrict__ flows,
                   const bf16_t* __restrict__ h_bf,
                   const bf16_t* __restrict__ wih_bf,
                   const bf16_t* __restrict__ whh_bf,
                   const float*  __restrict__ b_ih,
                   const float*  __restrict__ b_hh,
                   float*        __restrict__ h_fp,
                   bf16_t*       __restrict__ h_bf_out) {
  const int tile  = blockIdx.x * 8 + (threadIdx.x >> 5);
  const int ftile = tile & 7;
  const int rtile = tile >> 3;                // 0..511 row-tiles of 16 over BN=8192
  const int lane  = threadIdx.x & 31;
  const int col   = lane & 15;
  const int hi    = lane >> 4;

  const bf16_t* X = flows + (size_t)rtile * 16 * 512;
  const bf16_t* H = h_bf + (size_t)rtile * 16 * 128;

  v8f gi[3], gh[3];
#pragma unroll
  for (int g = 0; g < 3; ++g) {
    const float bi = b_ih[g * 128 + ftile * 16 + col];
    const float bh = b_hh[g * 128 + ftile * 16 + col];
#pragma unroll
    for (int v = 0; v < 8; ++v) { gi[g][v] = bi; gh[g][v] = bh; }
  }

#pragma unroll 4
  for (int k = 0; k < 512; k += 32) {
    v16bf a = load_a_frag(X, 512, k);
#pragma unroll
    for (int g = 0; g < 3; ++g) {
      v16bf bb = load_bt_frag(wih_bf + (size_t)(g * 128 + ftile * 16) * 512, 512, k);
      gi[g] = WMMA_BF16(a, bb, gi[g]);
    }
  }
#pragma unroll
  for (int k = 0; k < 128; k += 32) {
    v16bf a = load_a_frag(H, 128, k);
#pragma unroll
    for (int g = 0; g < 3; ++g) {
      v16bf bb = load_bt_frag(whh_bf + (size_t)(g * 128 + ftile * 16) * 128, 128, k);
      gh[g] = WMMA_BF16(a, bb, gh[g]);
    }
  }

  const int f = ftile * 16 + col;
#pragma unroll
  for (int v = 0; v < 8; ++v) {
    const size_t row = (size_t)rtile * 16 + hi * 8 + v;
    const size_t idx = row * 128 + f;
    const float hold = h_fp[idx];
    const float r = sigmoidf(gi[0][v] + gh[0][v]);
    const float z = sigmoidf(gi[1][v] + gh[1][v]);
    const float n = tanhf(gi[2][v] + r * gh[2][v]);
    const float hnew = (1.0f - z) * n + z * hold;
    h_fp[idx] = hnew;
    h_bf_out[idx] = (bf16_t)hnew;
  }
}

// ---------------------------------------------------------------------------
// Host launcher.  Workspace layout (bytes, all 256-aligned), ~88.7 MB total:
//   g_bf 0 / a_t 67108864 / flows 75497472 / hbf0 83886080 / hbf1 85983232
//   fcw_bf 88080384 / wih_bf 88211456 / whh_bf 88604672
// ---------------------------------------------------------------------------
extern "C" void kernel_launch(void* const* d_in, const int* in_sizes, int n_in,
                              void* d_out, int out_size, void* d_ws, size_t ws_size,
                              hipStream_t stream) {
  const float* feats = (const float*)d_in[0];   // (8,1024,128)
  const int*   graphs = (const int*)d_in[1];    // (8,4,1024,1024)
  const float* fc_w  = (const float*)d_in[2];   // (4,128,128)
  const float* fc_b  = (const float*)d_in[3];   // (4,128)
  const float* tw    = (const float*)d_in[4];   // (4,)
  const float* w_ih  = (const float*)d_in[5];   // (384,512)
  const float* w_hh  = (const float*)d_in[6];   // (384,128)
  const float* b_ih  = (const float*)d_in[7];   // (384,)
  const float* b_hh  = (const float*)d_in[8];   // (384,)
  float* out = (float*)d_out;                   // (8,1024,128) fp32, doubles as h

  char* ws = (char*)d_ws;
  bf16_t* g_bf   = (bf16_t*)(ws);
  bf16_t* a_t    = (bf16_t*)(ws + 67108864);
  bf16_t* flows  = (bf16_t*)(ws + 75497472);
  bf16_t* hbf0   = (bf16_t*)(ws + 83886080);
  bf16_t* hbf1   = (bf16_t*)(ws + 85983232);
  bf16_t* fcw_bf = (bf16_t*)(ws + 88080384);
  bf16_t* wih_bf = (bf16_t*)(ws + 88211456);
  bf16_t* whh_bf = (bf16_t*)(ws + 88604672);

  scale_graph_kernel<<<16384, 256, 0, stream>>>(graphs, tw, g_bf, 33554432);
  cvt_f32_bf16_kernel<<<(65536 + 255) / 256, 256, 0, stream>>>(fc_w, fcw_bf, 65536);
  cvt_f32_bf16_kernel<<<(196608 + 255) / 256, 256, 0, stream>>>(w_ih, wih_bf, 196608);
  cvt_f32_bf16_kernel<<<(49152 + 255) / 256, 256, 0, stream>>>(w_hh, whh_bf, 49152);
  cvt_copy_kernel<<<(1048576 + 255) / 256, 256, 0, stream>>>(feats, hbf0, out, 1048576);

  bf16_t* hin = hbf0;
  bf16_t* hout = hbf1;
  for (int s = 0; s < 5; ++s) {
    kA<<<2048, 256, 0, stream>>>(hin, fcw_bf, fc_b, a_t);
    kB<<<2048, 256, 0, stream>>>(g_bf, a_t, flows);
    kC<<<512, 256, 0, stream>>>(flows, hin, wih_bf, whh_bf, b_ih, b_hh, out, hout);
    bf16_t* t = hin; hin = hout; hout = t;
  }
  (void)in_sizes; (void)n_in; (void)out_size; (void)ws_size;
}
